// TransformerBlockQuantum_65481071403169
// MI455X (gfx1250) — matrix operand order repaired and checked by reading, not compiled, not run on hardware
//
#include <hip/hip_runtime.h>
#include <hip/hip_bf16.h>

typedef __attribute__((ext_vector_type(16))) __bf16 v16bf;
typedef __attribute__((ext_vector_type(8)))  __bf16 v8bf;
typedef __attribute__((ext_vector_type(8)))  float  v8f;

#define B_   8
#define S_   4096
#define E_   512
#define H_   64
#define NQ_  8
#define FFN_ 2048
#define M_   (B_ * S_)   // 32768 rows

// ---------- helpers ----------
__device__ __forceinline__ unsigned short f2bf(float f) {
    unsigned u = __float_as_uint(f);
    u += 0x7FFFu + ((u >> 16) & 1u);   // round-to-nearest-even
    return (unsigned short)(u >> 16);
}

// ---------- fp32 -> bf16 weight conversion ----------
__global__ __launch_bounds__(256)
void k_cvt_bf16(const float* __restrict__ src, unsigned short* __restrict__ dst, int n) {
    int i = blockIdx.x * 256 + threadIdx.x;
    if (i < n) dst[i] = f2bf(src[i]);
}

// ---------- quantum MHA, one block (64 threads) per token ----------
// Writes attention output in the torch-faithful scrambled row layout, bf16.
__global__ __launch_bounds__(64)
void k_attn(const float* __restrict__ x, const float* __restrict__ theta,
            unsigned short* __restrict__ A_scr) {
    const int t = blockIdx.x;          // token = b*S + s
    const int h = threadIdx.x;         // head 0..63
    const int b = t >> 12;             // S = 4096
    const int s = t & (S_ - 1);

    __shared__ float qs[H_][NQ_];
    __shared__ float th[NQ_];
    if (h < NQ_) th[h] = theta[h];
    __syncthreads();

    // qkv = cumprod(cos(x + theta)) for this head
    const float* xp = x + ((size_t)t << 9) + h * NQ_;
    float q[NQ_];
    float p = 1.0f;
#pragma unroll
    for (int i = 0; i < NQ_; ++i) {
        p *= cosf(xp[i] + th[i]);
        q[i] = p;
        qs[h][i] = p;
    }
    __syncthreads();

    const float scale = 0.35355339059327373f;   // 1/sqrt(8)
    // pass 1: row max of scores
    float mx = -1e30f;
    for (int g = 0; g < H_; ++g) {
        float d = 0.0f;
#pragma unroll
        for (int i = 0; i < NQ_; ++i) d += q[i] * qs[g][i];
        mx = fmaxf(mx, d * scale);
    }
    // pass 2: exp-sum fused with weighted accumulation
    float sum = 0.0f;
    float o[NQ_] = {0, 0, 0, 0, 0, 0, 0, 0};
    for (int g = 0; g < H_; ++g) {
        float d = 0.0f;
#pragma unroll
        for (int i = 0; i < NQ_; ++i) d += q[i] * qs[g][i];
        float e = __expf(d * scale - mx);
        sum += e;
#pragma unroll
        for (int i = 0; i < NQ_; ++i) o[i] += e * qs[g][i];
    }
    const float inv = 1.0f / sum;

    // scrambled destination: row = b*4096 + h*64 + s/64, col = (s%64)*8 + k
    const int m   = (b << 12) + (h << 6) + (s >> 6);
    const int col = (s & 63) << 3;
    union { unsigned short us[8]; uint4 v; } pk;
#pragma unroll
    for (int i = 0; i < NQ_; ++i) pk.us[i] = f2bf(o[i] * inv);
    *(uint4*)(A_scr + (((size_t)m) << 9) + col) = pk.v;
}

// ---------- bf16 WMMA GEMM:  C[M,N] (f32) = A[M,K] (bf16 rowmajor) * W[N,K]^T (bf16 rowmajor)
// block = 128 threads (4 waves); each wave -> 32(M) x 64(N) strip (8 accumulators);
// grid = (N/64, M/128). Per 32-K step: 12 b128 loads for 8 WMMAs (1.5 loads/WMMA).
__global__ __launch_bounds__(128)
void k_gemm_bf16(const unsigned short* __restrict__ A_, const unsigned short* __restrict__ W_,
                 float* __restrict__ C, int M, int N, int K) {
    const __bf16* A = (const __bf16*)A_;
    const __bf16* W = (const __bf16*)W_;

    const int lane  = threadIdx.x & 31;
    const int wave  = threadIdx.x >> 5;
    const int r15   = lane & 15;
    const int khalf = lane >> 4;                 // 0: lanes 0-15, 1: lanes 16-31
    const int m0 = (blockIdx.y * 4 + wave) * 32;
    const int n0 = blockIdx.x * 64;

    // A fragments: lane holds row m0+r15 (and m0+16+r15), K = khalf*16 .. +15 (contiguous)
    const __bf16* Ar0 = A + (size_t)(m0 + r15) * K + khalf * 16;
    const __bf16* Ar1 = Ar0 + (size_t)16 * K;
    // B fragments: lane holds col n0+16j+r15, K = khalf*16 .. +15 (contiguous in W's row)
    const __bf16* Wr  = W + (size_t)(n0 + r15) * K + khalf * 16;
    const size_t  wst = (size_t)16 * K;          // stride between N-fragments

    v8f acc00 = {}, acc01 = {}, acc02 = {}, acc03 = {};
    v8f acc10 = {}, acc11 = {}, acc12 = {}, acc13 = {};

    for (int k0 = 0; k0 < K; k0 += 32) {
        union { v16bf v; v8bf h[2]; } a0, a1, b0, b1, b2, b3;
        a0.h[0] = *(const v8bf*)(Ar0 + k0);
        a0.h[1] = *(const v8bf*)(Ar0 + k0 + 8);
        a1.h[0] = *(const v8bf*)(Ar1 + k0);
        a1.h[1] = *(const v8bf*)(Ar1 + k0 + 8);
        const __bf16* w = Wr + k0;
        b0.h[0] = *(const v8bf*)(w);               b0.h[1] = *(const v8bf*)(w + 8);
        b1.h[0] = *(const v8bf*)(w + wst);         b1.h[1] = *(const v8bf*)(w + wst + 8);
        b2.h[0] = *(const v8bf*)(w + 2 * wst);     b2.h[1] = *(const v8bf*)(w + 2 * wst + 8);
        b3.h[0] = *(const v8bf*)(w + 3 * wst);     b3.h[1] = *(const v8bf*)(w + 3 * wst + 8);

        acc00 = __builtin_amdgcn_wmma_f32_16x16x32_bf16(false, a0.v, false, b0.v, (short)0, acc00, false, false);
        acc01 = __builtin_amdgcn_wmma_f32_16x16x32_bf16(false, a0.v, false, b1.v, (short)0, acc01, false, false);
        acc02 = __builtin_amdgcn_wmma_f32_16x16x32_bf16(false, a0.v, false, b2.v, (short)0, acc02, false, false);
        acc03 = __builtin_amdgcn_wmma_f32_16x16x32_bf16(false, a0.v, false, b3.v, (short)0, acc03, false, false);
        acc10 = __builtin_amdgcn_wmma_f32_16x16x32_bf16(false, a1.v, false, b0.v, (short)0, acc10, false, false);
        acc11 = __builtin_amdgcn_wmma_f32_16x16x32_bf16(false, a1.v, false, b1.v, (short)0, acc11, false, false);
        acc12 = __builtin_amdgcn_wmma_f32_16x16x32_bf16(false, a1.v, false, b2.v, (short)0, acc12, false, false);
        acc13 = __builtin_amdgcn_wmma_f32_16x16x32_bf16(false, a1.v, false, b3.v, (short)0, acc13, false, false);
    }

    // D layout: VGPR r -> row m0 + r + khalf*8 (+16 for second strip), col n0 + 16j + r15
#pragma unroll
    for (int r = 0; r < 8; ++r) {
        float* p0 = C + (size_t)(m0 + r + khalf * 8) * N + n0 + r15;
        p0[0]  = acc00[r];
        p0[16] = acc01[r];
        p0[32] = acc02[r];
        p0[48] = acc03[r];
        float* p1 = p0 + (size_t)16 * N;
        p1[0]  = acc10[r];
        p1[16] = acc11[r];
        p1[32] = acc12[r];
        p1[48] = acc13[r];
    }
}

// ---------- LN1 + quantum-FFN front + h = relu(q @ w1^T); one block (256 thr) per row ----------
__global__ __launch_bounds__(256)
void k_ln1_ffn(const float* __restrict__ x, const float* __restrict__ y1,
               const float* __restrict__ g, const float* __restrict__ bb,
               const float* __restrict__ th_ffn, const float* __restrict__ w1,
               float* __restrict__ x1, unsigned short* __restrict__ Hb) {
    const int m = blockIdx.x, tid = threadIdx.x;
    __shared__ float red[256];
    __shared__ float qsh[NQ_];
    const size_t base = (size_t)m * E_;

    float v0 = x[base + tid]       + y1[base + tid];
    float v1 = x[base + tid + 256] + y1[base + tid + 256];

    red[tid] = v0 + v1;
    __syncthreads();
    for (int off = 128; off > 0; off >>= 1) { if (tid < off) red[tid] += red[tid + off]; __syncthreads(); }
    const float mu = red[0] * (1.0f / (float)E_);
    __syncthreads();

    const float d0 = v0 - mu, d1 = v1 - mu;
    red[tid] = d0 * d0 + d1 * d1;
    __syncthreads();
    for (int off = 128; off > 0; off >>= 1) { if (tid < off) red[tid] += red[tid + off]; __syncthreads(); }
    const float rstd = rsqrtf(red[0] * (1.0f / (float)E_) + 1e-5f);

    const float o0 = d0 * rstd * g[tid]       + bb[tid];
    const float o1 = d1 * rstd * g[tid + 256] + bb[tid + 256];
    x1[base + tid]       = o0;
    x1[base + tid + 256] = o1;

    if (tid < NQ_) qsh[tid] = cosf(th_ffn[tid]) * cosf(o0);   // x1[..., :8]
    __syncthreads();

    const size_t hbase = (size_t)m * FFN_;
    for (int f = tid; f < FFN_; f += 256) {
        float acc = 0.0f;
#pragma unroll
        for (int k = 0; k < NQ_; ++k) acc += qsh[k] * w1[f * NQ_ + k];
        Hb[hbase + f] = f2bf(fmaxf(acc, 0.0f));
    }
}

// ---------- LN2 -> final output ----------
__global__ __launch_bounds__(256)
void k_ln2(const float* __restrict__ x1, const float* __restrict__ y2,
           const float* __restrict__ g, const float* __restrict__ bb,
           float* __restrict__ out) {
    const int m = blockIdx.x, tid = threadIdx.x;
    __shared__ float red[256];
    const size_t base = (size_t)m * E_;

    float v0 = x1[base + tid]       + y2[base + tid];
    float v1 = x1[base + tid + 256] + y2[base + tid + 256];

    red[tid] = v0 + v1;
    __syncthreads();
    for (int off = 128; off > 0; off >>= 1) { if (tid < off) red[tid] += red[tid + off]; __syncthreads(); }
    const float mu = red[0] * (1.0f / (float)E_);
    __syncthreads();

    const float d0 = v0 - mu, d1 = v1 - mu;
    red[tid] = d0 * d0 + d1 * d1;
    __syncthreads();
    for (int off = 128; off > 0; off >>= 1) { if (tid < off) red[tid] += red[tid + off]; __syncthreads(); }
    const float rstd = rsqrtf(red[0] * (1.0f / (float)E_) + 1e-5f);

    out[base + tid]       = d0 * rstd * g[tid]       + bb[tid];
    out[base + tid + 256] = d1 * rstd * g[tid + 256] + bb[tid + 256];
}

extern "C" void kernel_launch(void* const* d_in, const int* in_sizes, int n_in,
                              void* d_out, int out_size, void* d_ws, size_t ws_size,
                              hipStream_t stream) {
    const float* x       = (const float*)d_in[0];
    const float* th_attn = (const float*)d_in[1];
    const float* w_comb  = (const float*)d_in[2];
    const float* th_ffn  = (const float*)d_in[3];
    const float* w1      = (const float*)d_in[4];
    const float* w2      = (const float*)d_in[5];
    const float* ln1_g   = (const float*)d_in[6];
    const float* ln1_b   = (const float*)d_in[7];
    const float* ln2_g   = (const float*)d_in[8];
    const float* ln2_b   = (const float*)d_in[9];
    float* out = (float*)d_out;

    char* ws = (char*)d_ws;
    size_t off = 0;
    auto take = [&](size_t bytes) -> char* {
        char* p = ws + off;
        off = (off + bytes + 255) & ~(size_t)255;
        return p;
    };
    unsigned short* A_scr = (unsigned short*)take((size_t)M_ * E_   * 2);  // scrambled attn out, bf16
    unsigned short* WcB   = (unsigned short*)take((size_t)E_  * E_   * 2);  // w_combine bf16
    unsigned short* W2B   = (unsigned short*)take((size_t)E_  * FFN_ * 2);  // w2 bf16
    float*          Y     = (float*)         take((size_t)M_ * E_   * 4);  // GEMM1 out, reused for GEMM2 out
    float*          X1    = (float*)         take((size_t)M_ * E_   * 4);  // post-LN1 activations
    unsigned short* Hb    = (unsigned short*)take((size_t)M_ * FFN_ * 2);  // relu hidden, bf16

    k_cvt_bf16<<<(E_ * E_   + 255) / 256, 256, 0, stream>>>(w_comb, WcB, E_ * E_);
    k_cvt_bf16<<<(E_ * FFN_ + 255) / 256, 256, 0, stream>>>(w2,     W2B, E_ * FFN_);

    k_attn<<<M_, 64, 0, stream>>>(x, th_attn, A_scr);

    // attn_out = A_scr @ w_combine^T   (M=32768, N=512, K=512)
    k_gemm_bf16<<<dim3(E_ / 64, M_ / 128), 128, 0, stream>>>(A_scr, WcB, Y, M_, E_, E_);

    k_ln1_ffn<<<M_, 256, 0, stream>>>(x, Y, ln1_g, ln1_b, th_ffn, w1, X1, Hb);

    // ffn_out = Hb @ w2^T   (M=32768, N=512, K=2048)
    k_gemm_bf16<<<dim3(E_ / 64, M_ / 128), 128, 0, stream>>>(Hb, W2B, Y, M_, E_, FFN_);

    k_ln2<<<M_, 256, 0, stream>>>(X1, Y, ln2_g, ln2_b, out);
}
